// CSDTI_36850819400267
// MI455X (gfx1250) — matrix operand order repaired and checked by reading, not compiled, not run on hardware
//
#include <hip/hip_runtime.h>

typedef __attribute__((ext_vector_type(16))) _Float16 v16h;
typedef __attribute__((ext_vector_type(8)))  _Float16 v8h;
typedef __attribute__((ext_vector_type(8)))  float    v8f;

#define N_NODES 12800
#define N_EDGES 102400
#define NG      128
#define PLEN    1200
#define EMBD    512
#define VOCAB   26

// ---------------------------------------------------------------------------
// WMMA GEMM, f16 operands / f32 accum:
//   C[b] = act(A[b][M,K] @ B[K,N] + bias[N]),  B given pre-transposed BT[N][K].
// K must be a multiple of 32 (pre-padded). One wave per 16x16 tile.
// Fragments are contiguous b128 loads per the gfx1250 VGPR striping:
//   A lane: rows m0+(lane&15); halves = [k0+8h..+7] ++ [k0+16+8h..+7], h=lane>>4
//   B lane: col n0+(lane&15); halves = [k0+8h..+7] ++ [k0+16+8h..+7]
// ---------------------------------------------------------------------------
__global__ void gemm_wmma_f16(const _Float16* __restrict__ A, const _Float16* __restrict__ BT,
                              const float* __restrict__ bias,
                              float* __restrict__ C, _Float16* __restrict__ C16,
                              int M, int N, int K, int lda, int ldc,
                              long sA, long sC, int relu)
{
    int lane = threadIdx.x & 31;
    int wave = threadIdx.x >> 5;
    int tilesN = N >> 4, tilesM = M >> 4;
    int tile = blockIdx.x * (blockDim.x >> 5) + wave;
    if (tile >= tilesM * tilesN) return;
    long b = blockIdx.y;
    int m0 = (tile / tilesN) << 4;
    int n0 = (tile % tilesN) << 4;
    int half = lane >> 4, l15 = lane & 15;
    const _Float16* pa = A + b * sA + (long)(m0 + l15) * lda + half * 8;
    const _Float16* pb = BT + (long)(n0 + l15) * K + half * 8;
    v8f acc = {};
    for (int k0 = 0; k0 < K; k0 += 32) {
        v8h alo = *(const v8h*)(pa + k0);
        v8h ahi = *(const v8h*)(pa + k0 + 16);
        v8h blo = *(const v8h*)(pb + k0);
        v8h bhi = *(const v8h*)(pb + k0 + 16);
        v16h a  = __builtin_shufflevector(alo, ahi, 0,1,2,3,4,5,6,7,8,9,10,11,12,13,14,15);
        v16h bf = __builtin_shufflevector(blo, bhi, 0,1,2,3,4,5,6,7,8,9,10,11,12,13,14,15);
        acc = __builtin_amdgcn_wmma_f32_16x16x32_f16(false, a, false, bf,
                                                     (short)0, acc, false, false);
    }
#pragma unroll
    for (int r = 0; r < 8; ++r) {
        int m = m0 + r + half * 8;
        int n = n0 + l15;
        float v = acc[r];
        if (bias) v += bias[n];
        if (relu) v = fmaxf(v, 0.0f);
        if (C)   C[b * sC + (long)m * ldc + n] = v;
        if (C16) C16[b * sC + (long)m * ldc + n] = (_Float16)v;
    }
}

// ---------------------------------------------------------------------------
// Protein bin reduction: Wsum[b][o][v] = sum_{l: tgt[b,l]==v} W[o,l]; slot 31 = bias
// WT is convxt_w transposed [1200][64] (coalesced across o). Bins in LDS (pad 33).
// ---------------------------------------------------------------------------
__global__ void protein_bins(const float* __restrict__ WT, const int* __restrict__ target,
                             const float* __restrict__ cb, _Float16* __restrict__ Wsum)
{
    __shared__ int   s_tgt[PLEN];
    __shared__ float bins[64 * 33];
    int b = blockIdx.x;
    int o = threadIdx.x;                       // blockDim.x == 64
    for (int i = o; i < PLEN; i += 64) s_tgt[i] = target[b * PLEN + i];
    for (int v = 0; v < 32; ++v) bins[o * 33 + v] = 0.0f;
    __syncthreads();
    for (int l = 0; l < PLEN; ++l)
        bins[o * 33 + s_tgt[l]] += WT[l * 64 + o];
    for (int v = 0; v < 32; ++v) {
        float val = (v == 31) ? cb[o] : bins[o * 33 + v];
        Wsum[((long)b * 64 + o) * 32 + v] = (_Float16)val;
    }
}

// ---------------------------------------------------------------------------
// GLU conv layer, h layout [B, L=64, C=64] channel-last, f32 in/out.
// LDS tile has one zero row above/below -> branch-free shifted A fragments.
// cwT16 layout [layer][t][o=128][i=64] -> contiguous B fragments.
// ---------------------------------------------------------------------------
__global__ void conv_glu_layer(const float* __restrict__ h_in,
                               const _Float16* __restrict__ cwT16,
                               const float* __restrict__ bias,   // [4][128]
                               float* __restrict__ h_out, int layer)
{
    __shared__ _Float16 s_h[66 * 64];
    __shared__ float    s_c[64 * 128];
    int b = blockIdx.x;
    const float* hb = h_in + (long)b * 4096;
    for (int i = threadIdx.x; i < 64; i += blockDim.x) {
        s_h[i] = (_Float16)0.0f;
        s_h[65 * 64 + i] = (_Float16)0.0f;
    }
    for (int i = threadIdx.x; i < 4096; i += blockDim.x) s_h[64 + i] = (_Float16)hb[i];
    __syncthreads();

    int lane = threadIdx.x & 31, wave = threadIdx.x >> 5;
    int half = lane >> 4, l15 = lane & 15;
    const _Float16* wl = cwT16 + layer * 3 * 128 * 64;
    for (int it = 0; it < 4; ++it) {
        int tile = wave + it * 8;              // 32 tiles: 4 m(L) x 8 n(O)
        int m0 = (tile >> 3) << 4;
        int n0 = (tile & 7) << 4;
        v8f acc = {};
        for (int t = 0; t < 3; ++t) {
            int lp = m0 + l15 + t;             // padded row index (l + t - 1 + 1)
            const _Float16* ps = s_h + lp * 64 + half * 8;
            const _Float16* pw = wl + t * 128 * 64 + (n0 + l15) * 64 + half * 8;
#pragma unroll
            for (int k0 = 0; k0 < 64; k0 += 32) {
                v8h alo = *(const v8h*)(ps + k0);
                v8h ahi = *(const v8h*)(ps + k0 + 16);
                v8h blo = *(const v8h*)(pw + k0);
                v8h bhi = *(const v8h*)(pw + k0 + 16);
                v16h a  = __builtin_shufflevector(alo, ahi, 0,1,2,3,4,5,6,7,8,9,10,11,12,13,14,15);
                v16h bf = __builtin_shufflevector(blo, bhi, 0,1,2,3,4,5,6,7,8,9,10,11,12,13,14,15);
                acc = __builtin_amdgcn_wmma_f32_16x16x32_f16(false, a, false, bf,
                                                             (short)0, acc, false, false);
            }
        }
#pragma unroll
        for (int r = 0; r < 8; ++r) {
            int l = m0 + r + half * 8, o = n0 + l15;
            s_c[l * 128 + o] = acc[r] + bias[layer * 128 + o];
        }
    }
    __syncthreads();
    for (int idx = threadIdx.x; idx < 4096; idx += blockDim.x) {
        int l = idx >> 6, c = idx & 63;
        float av = s_c[l * 128 + c];
        float gv = s_c[l * 128 + 64 + c];
        h_out[(long)b * 4096 + idx] = av * (1.0f / (1.0f + __expf(-gv)))
                                    + hb[idx] * 0.70710678118654752f;
    }
}

// ---------------------------------------------------------------------------
// data prep / sparse / misc helpers
// ---------------------------------------------------------------------------
__global__ void fill_f32(float* p, float v, int n) {
    int i = blockIdx.x * blockDim.x + threadIdx.x;
    if (i < n) p[i] = v;
}
__global__ void cvt_f16(const float* in, _Float16* out, int n) {
    int i = blockIdx.x * blockDim.x + threadIdx.x;
    if (i < n) out[i] = (_Float16)in[i];
}
// weight [K,N] f32 -> [N,Kpad] f16 (zero-padded)
__global__ void wT16(const float* w, _Float16* out, int K, int N, int Kpad) {
    int idx = blockIdx.x * blockDim.x + threadIdx.x;
    if (idx >= N * Kpad) return;
    int n = idx / Kpad, k = idx % Kpad;
    out[idx] = (k < K) ? (_Float16)w[(long)k * N + n] : (_Float16)0.0f;
}
// x [12800,22] f32 -> [12800,32] f16 zero-padded
__global__ void pad_x16(const float* x, _Float16* out) {
    int idx = blockIdx.x * blockDim.x + threadIdx.x;
    if (idx >= N_NODES * 32) return;
    int i = idx >> 5, c = idx & 31;
    out[idx] = (c < 22) ? (_Float16)x[i * 22 + c] : (_Float16)0.0f;
}
// convxt_w [64,1200] -> WT [1200,64] f32
__global__ void transpose_w(const float* w, float* out) {
    int idx = blockIdx.x * blockDim.x + threadIdx.x;
    if (idx >= PLEN * 64) return;
    int l = idx / 64, o = idx % 64;
    out[idx] = w[o * PLEN + l];
}
// emb [26,512] -> embT [512,32] f16, pad rows 26..30 = 0, row 31 = 1 (bias row)
__global__ void embT16(const float* emb, _Float16* out) {
    int idx = blockIdx.x * blockDim.x + threadIdx.x;
    if (idx >= EMBD * 32) return;
    int n = idx / 32, v = idx % 32;
    float val = (v < VOCAB) ? emb[v * EMBD + n] : (v == 31 ? 1.0f : 0.0f);
    out[idx] = (_Float16)val;
}
// enc_conv_w [4][128][64][3] -> [4][3][128][64] f16
__global__ void conv_wT16(const float* w, _Float16* out) {
    int idx = blockIdx.x * blockDim.x + threadIdx.x;
    if (idx >= 4 * 3 * 128 * 64) return;
    int i = idx & 63, r = idx >> 6;
    int o = r & 127, r2 = r >> 7;
    int t = r2 % 3, L = r2 / 3;
    out[idx] = (_Float16)w[((long)(L * 128 + o) * 64 + i) * 3 + t];
}
__global__ void deg_accum(const int* dst, float* deg, int nE) {
    int e = blockIdx.x * blockDim.x + threadIdx.x;
    if (e < nE) atomicAdd(&deg[dst[e]], 1.0f);
}
__global__ void to_rsqrt(float* p, int n) {
    int i = blockIdx.x * blockDim.x + threadIdx.x;
    if (i < n) p[i] = rsqrtf(p[i]);
}
__global__ void gcn_self_init(const float* hw, const float* dinv, const float* bias,
                              float* out, int nNodes) {
    int idx = blockIdx.x * blockDim.x + threadIdx.x;
    if (idx >= nNodes * 32) return;
    int i = idx >> 5, c = idx & 31;
    float d = dinv[i];
    out[idx] = bias[c] + hw[idx] * d * d;
}
__global__ void gcn_edge(const int* src, const int* dst, const float* hw,
                         const float* dinv, float* out, int nE) {
    int idx = blockIdx.x * blockDim.x + threadIdx.x;
    if (idx >= nE * 32) return;
    int e = idx >> 5, c = idx & 31;
    int s = src[e], d = dst[e];
    atomicAdd(&out[d * 32 + c], hw[s * 32 + c] * dinv[s] * dinv[d]);
}
__global__ void relu_concat2(const float* a, const float* b, float* x12,
                             _Float16* x12h, int nNodes) {
    int idx = blockIdx.x * blockDim.x + threadIdx.x;
    if (idx >= nNodes * 32) return;
    int i = idx >> 5, c = idx & 31;
    float va = fmaxf(a[idx], 0.0f), vb = fmaxf(b[idx], 0.0f);
    x12[i * 64 + c] = va;
    x12[i * 64 + 32 + c] = vb;
    if (x12h) {
        x12h[i * 64 + c] = (_Float16)va;
        x12h[i * 64 + 32 + c] = (_Float16)vb;
    }
}
__global__ void pool_scatter(const float* x12, const int* batch, float* pooled,
                             float* cnt, int nNodes) {
    int idx = blockIdx.x * blockDim.x + threadIdx.x;
    if (idx >= nNodes * 64) return;
    int i = idx >> 6, c = idx & 63;
    int g = batch[i];
    atomicAdd(&pooled[g * 64 + c], x12[idx]);
    if (c == 0) atomicAdd(&cnt[g], 1.0f);
}
__global__ void pool_div(float* pooled, const float* cnt) {
    int idx = blockIdx.x * blockDim.x + threadIdx.x;
    if (idx >= NG * 64) return;
    pooled[idx] /= fmaxf(cnt[idx >> 6], 1.0f);
}
__global__ void layernorm64(const float* h, const float* g, const float* be, _Float16* out) {
    int idx = blockIdx.x * blockDim.x + threadIdx.x;   // (b,l)
    if (idx >= NG * 64) return;
    const float* row = h + (long)idx * 64;
    float mu = 0.f;
    for (int c = 0; c < 64; ++c) mu += row[c];
    mu *= (1.0f / 64.0f);
    float var = 0.f;
    for (int c = 0; c < 64; ++c) { float d = row[c] - mu; var += d * d; }
    var *= (1.0f / 64.0f);
    float inv = rsqrtf(var + 1e-5f);
    _Float16* orow = out + (long)idx * 64;
    for (int c = 0; c < 64; ++c) orow[c] = (_Float16)((row[c] - mu) * inv * g[c] + be[c]);
}
// rank-1 softmax attention; one wave per (b,i), wave32 shuffles
__global__ void attention_rank1(const float* Q, const float* K, const float* V, _Float16* att) {
    int lane = threadIdx.x & 31;
    int wid = (blockIdx.x * blockDim.x + threadIdx.x) >> 5;
    if (wid >= NG * 128) return;
    int b = wid >> 7, i = wid & 127;
    float qi = Q[b * 128 + i] * 0.08838834764831845f;
    float l[4];
    float m = -1e30f;
#pragma unroll
    for (int t = 0; t < 4; ++t) { int j = lane + t * 32; l[t] = qi * K[b * 128 + j]; m = fmaxf(m, l[t]); }
    for (int o = 16; o >= 1; o >>= 1) m = fmaxf(m, __shfl_xor(m, o, 32));
    float se = 0.f, acc = 0.f;
#pragma unroll
    for (int t = 0; t < 4; ++t) {
        int j = lane + t * 32;
        float e = __expf(l[t] - m);
        se += e; acc += e * V[b * 128 + j];
    }
    for (int o = 16; o >= 1; o >>= 1) { se += __shfl_xor(se, o, 32); acc += __shfl_xor(acc, o, 32); }
    if (lane == 0) att[b * 128 + i] = (_Float16)(acc / se);
}
__global__ void concat3(const float* xd, const float* att, const float* xt, _Float16* f) {
    int idx = blockIdx.x * blockDim.x + threadIdx.x;
    if (idx >= NG * 384) return;
    int b = idx / 384, c = idx % 384;
    float v = (c < 128) ? xd[b * 128 + c]
            : (c < 256) ? att[b * 128 + (c - 128)]
                        : xt[b * 128 + (c - 256)];
    f[idx] = (_Float16)v;
}
__global__ void final_m4(const float* h3, const float* w, const float* b, float* out) {
    int i = blockIdx.x * blockDim.x + threadIdx.x;
    if (i >= NG) return;
    float s = b[0];
    for (int k = 0; k < 256; ++k) s += h3[i * 256 + k] * w[k];
    out[i] = s;
}

// ---------------------------------------------------------------------------
static inline void launch_gemm(const _Float16* A, const _Float16* BT, const float* bias,
                               float* C, _Float16* C16, int M, int N, int K,
                               int lda, int ldc, long sA, long sC, int relu, int batch,
                               hipStream_t s)
{
    int tiles = (M / 16) * (N / 16);
    dim3 grid((tiles + 3) / 4, batch);
    gemm_wmma_f16<<<grid, 128, 0, s>>>(A, BT, bias, C, C16, M, N, K, lda, ldc, sA, sC, relu);
}
static inline dim3 ez(int n) { return dim3((n + 255) / 256); }

extern "C" void kernel_launch(void* const* d_in, const int* in_sizes, int n_in,
                              void* d_out, int out_size, void* d_ws, size_t ws_size,
                              hipStream_t stream)
{
    const float* x         = (const float*)d_in[0];
    const float* w11       = (const float*)d_in[1];
    const float* b11       = (const float*)d_in[2];
    const float* w12       = (const float*)d_in[3];
    const float* b12       = (const float*)d_in[4];
    const float* w21       = (const float*)d_in[5];
    const float* b21       = (const float*)d_in[6];
    const float* w22       = (const float*)d_in[7];
    const float* b22       = (const float*)d_in[8];
    const float* fc1_xd_w  = (const float*)d_in[9];
    const float* fc1_xd_b  = (const float*)d_in[10];
    const float* emb       = (const float*)d_in[11];
    const float* convxt_w  = (const float*)d_in[12];
    const float* convxt_b  = (const float*)d_in[13];
    const float* enc_fc_w  = (const float*)d_in[14];
    const float* enc_fc_b  = (const float*)d_in[15];
    const float* enc_conv_w= (const float*)d_in[16];
    const float* enc_conv_b= (const float*)d_in[17];
    const float* ln_g      = (const float*)d_in[18];
    const float* ln_b      = (const float*)d_in[19];
    const float* fc1_xt_w  = (const float*)d_in[20];
    const float* fc1_xt_b  = (const float*)d_in[21];
    const float* fq_w      = (const float*)d_in[22];
    const float* fq_b      = (const float*)d_in[23];
    const float* fk_w      = (const float*)d_in[24];
    const float* fk_b      = (const float*)d_in[25];
    const float* fv_w      = (const float*)d_in[26];
    const float* fv_b      = (const float*)d_in[27];
    const float* fo_w      = (const float*)d_in[28];
    const float* fo_b      = (const float*)d_in[29];
    const float* m1_w      = (const float*)d_in[30];
    const float* m1_b      = (const float*)d_in[31];
    const float* m2_w      = (const float*)d_in[32];
    const float* m2_b      = (const float*)d_in[33];
    const float* m3_w      = (const float*)d_in[34];
    const float* m3_b      = (const float*)d_in[35];
    const float* m4_w      = (const float*)d_in[36];
    const float* m4_b      = (const float*)d_in[37];
    const int*   ei1       = (const int*)d_in[38];
    const int*   ei2       = (const int*)d_in[39];
    const int*   batch     = (const int*)d_in[40];
    const int*   target    = (const int*)d_in[41];
    float* out = (float*)d_out;
    (void)ws_size; (void)in_sizes; (void)n_in; (void)out_size;

    char* ws = (char*)d_ws;
    size_t off = 0;
    auto alloc = [&](size_t bytes) -> void* {
        void* p = ws + off;
        off = (off + bytes + 255) & ~(size_t)255;
        return p;
    };
    // f32 buffers
    float* dinv1  = (float*)alloc(N_NODES * 4);
    float* dinv2  = (float*)alloc(N_NODES * 4);
    float* hw1    = (float*)alloc((size_t)N_NODES * 32 * 4);
    float* hw2    = (float*)alloc((size_t)N_NODES * 32 * 4);
    float* g1     = (float*)alloc((size_t)N_NODES * 32 * 4);
    float* g2     = (float*)alloc((size_t)N_NODES * 32 * 4);
    float* x12    = (float*)alloc((size_t)N_NODES * 64 * 4);
    float* x12b   = (float*)alloc((size_t)N_NODES * 64 * 4);
    float* pooled = (float*)alloc(NG * 64 * 4);
    float* cnt    = (float*)alloc(NG * 4);
    float* xd     = (float*)alloc(NG * 128 * 4);
    float* wtr    = (float*)alloc((size_t)PLEN * 64 * 4);
    float* hA     = (float*)alloc((size_t)NG * 4096 * 4);
    float* hB     = (float*)alloc((size_t)NG * 4096 * 4);
    float* xt128  = (float*)alloc(NG * 128 * 4);
    float* Qb     = (float*)alloc(NG * 128 * 4);
    float* Kb     = (float*)alloc(NG * 128 * 4);
    float* Vb     = (float*)alloc(NG * 128 * 4);
    float* atto   = (float*)alloc(NG * 128 * 4);
    float* h3     = (float*)alloc(NG * 256 * 4);
    // f16 buffers
    _Float16* x16     = (_Float16*)alloc((size_t)N_NODES * 32 * 2);
    _Float16* x12h    = (_Float16*)alloc((size_t)N_NODES * 64 * 2);
    _Float16* pooled16= (_Float16*)alloc(NG * 64 * 2);
    _Float16* xd16    = (_Float16*)alloc(NG * 128 * 2);
    _Float16* Wsum16  = (_Float16*)alloc((size_t)NG * 64 * 32 * 2);
    _Float16* xt51216 = (_Float16*)alloc((size_t)NG * 64 * EMBD * 2);
    _Float16* ln16    = (_Float16*)alloc((size_t)NG * 4096 * 2);
    _Float16* xt12816 = (_Float16*)alloc(NG * 128 * 2);
    _Float16* att16   = (_Float16*)alloc(NG * 128 * 2);
    _Float16* f38416  = (_Float16*)alloc(NG * 384 * 2);
    _Float16* h116    = (_Float16*)alloc(NG * 1024 * 2);
    _Float16* h216    = (_Float16*)alloc(NG * 1024 * 2);
    // transposed/padded f16 weights
    _Float16* w11T   = (_Float16*)alloc(32 * 32 * 2);
    _Float16* w12T   = (_Float16*)alloc(32 * 32 * 2);
    _Float16* w21T   = (_Float16*)alloc(32 * 64 * 2);
    _Float16* w22T   = (_Float16*)alloc(32 * 64 * 2);
    _Float16* fcxdT  = (_Float16*)alloc(128 * 64 * 2);
    _Float16* encfcT = (_Float16*)alloc((size_t)64 * 512 * 2);
    _Float16* fcxtT  = (_Float16*)alloc((size_t)128 * 4096 * 2);
    _Float16* fqT    = (_Float16*)alloc(128 * 128 * 2);
    _Float16* fkT    = (_Float16*)alloc(128 * 128 * 2);
    _Float16* fvT    = (_Float16*)alloc(128 * 128 * 2);
    _Float16* foT    = (_Float16*)alloc(128 * 128 * 2);
    _Float16* m1T    = (_Float16*)alloc((size_t)1024 * 384 * 2);
    _Float16* m2T    = (_Float16*)alloc((size_t)1024 * 1024 * 2);
    _Float16* m3T    = (_Float16*)alloc((size_t)256 * 1024 * 2);
    _Float16* cwT16  = (_Float16*)alloc((size_t)4 * 3 * 128 * 64 * 2);
    _Float16* embpT  = (_Float16*)alloc(EMBD * 32 * 2);

    const int* src1 = ei1, * dst1 = ei1 + N_EDGES;
    const int* src2 = ei2, * dst2 = ei2 + N_EDGES;

    // ---- weight prep (f16, transposed, K padded to /32) ----
    pad_x16<<<ez(N_NODES * 32), 256, 0, stream>>>(x, x16);
    wT16<<<ez(32 * 32), 256, 0, stream>>>(w11, w11T, 22, 32, 32);
    wT16<<<ez(32 * 32), 256, 0, stream>>>(w12, w12T, 22, 32, 32);
    wT16<<<ez(32 * 64), 256, 0, stream>>>(w21, w21T, 64, 32, 64);
    wT16<<<ez(32 * 64), 256, 0, stream>>>(w22, w22T, 64, 32, 64);
    wT16<<<ez(128 * 64), 256, 0, stream>>>(fc1_xd_w, fcxdT, 64, 128, 64);
    wT16<<<ez(64 * 512), 256, 0, stream>>>(enc_fc_w, encfcT, 512, 64, 512);
    wT16<<<ez(128 * 4096), 256, 0, stream>>>(fc1_xt_w, fcxtT, 4096, 128, 4096);
    wT16<<<ez(128 * 128), 256, 0, stream>>>(fq_w, fqT, 128, 128, 128);
    wT16<<<ez(128 * 128), 256, 0, stream>>>(fk_w, fkT, 128, 128, 128);
    wT16<<<ez(128 * 128), 256, 0, stream>>>(fv_w, fvT, 128, 128, 128);
    wT16<<<ez(128 * 128), 256, 0, stream>>>(fo_w, foT, 128, 128, 128);
    wT16<<<ez(1024 * 384), 256, 0, stream>>>(m1_w, m1T, 384, 1024, 384);
    wT16<<<ez(1024 * 1024), 256, 0, stream>>>(m2_w, m2T, 1024, 1024, 1024);
    wT16<<<ez(256 * 1024), 256, 0, stream>>>(m3_w, m3T, 1024, 256, 1024);
    conv_wT16<<<ez(4 * 3 * 128 * 64), 256, 0, stream>>>(enc_conv_w, cwT16);
    transpose_w<<<ez(PLEN * 64), 256, 0, stream>>>(convxt_w, wtr);
    embT16<<<ez(EMBD * 32), 256, 0, stream>>>(emb, embpT);

    // ---------------- GCN branch ----------------
    fill_f32<<<ez(N_NODES), 256, 0, stream>>>(dinv1, 1.0f, N_NODES);
    deg_accum<<<ez(N_EDGES), 256, 0, stream>>>(dst1, dinv1, N_EDGES);
    to_rsqrt<<<ez(N_NODES), 256, 0, stream>>>(dinv1, N_NODES);
    fill_f32<<<ez(N_NODES), 256, 0, stream>>>(dinv2, 1.0f, N_NODES);
    deg_accum<<<ez(N_EDGES), 256, 0, stream>>>(dst2, dinv2, N_EDGES);
    to_rsqrt<<<ez(N_NODES), 256, 0, stream>>>(dinv2, N_NODES);

    launch_gemm(x16, w11T, nullptr, hw1, nullptr, N_NODES, 32, 32, 32, 32, 0, 0, 0, 1, stream);
    launch_gemm(x16, w12T, nullptr, hw2, nullptr, N_NODES, 32, 32, 32, 32, 0, 0, 0, 1, stream);
    gcn_self_init<<<ez(N_NODES * 32), 256, 0, stream>>>(hw1, dinv1, b11, g1, N_NODES);
    gcn_edge<<<ez(N_EDGES * 32), 256, 0, stream>>>(src1, dst1, hw1, dinv1, g1, N_EDGES);
    gcn_self_init<<<ez(N_NODES * 32), 256, 0, stream>>>(hw2, dinv2, b12, g2, N_NODES);
    gcn_edge<<<ez(N_EDGES * 32), 256, 0, stream>>>(src2, dst2, hw2, dinv2, g2, N_EDGES);
    relu_concat2<<<ez(N_NODES * 32), 256, 0, stream>>>(g1, g2, x12, x12h, N_NODES);

    launch_gemm(x12h, w21T, nullptr, hw1, nullptr, N_NODES, 32, 64, 64, 32, 0, 0, 0, 1, stream);
    launch_gemm(x12h, w22T, nullptr, hw2, nullptr, N_NODES, 32, 64, 64, 32, 0, 0, 0, 1, stream);
    gcn_self_init<<<ez(N_NODES * 32), 256, 0, stream>>>(hw1, dinv1, b21, g1, N_NODES);
    gcn_edge<<<ez(N_EDGES * 32), 256, 0, stream>>>(src1, dst1, hw1, dinv1, g1, N_EDGES);
    gcn_self_init<<<ez(N_NODES * 32), 256, 0, stream>>>(hw2, dinv2, b22, g2, N_NODES);
    gcn_edge<<<ez(N_EDGES * 32), 256, 0, stream>>>(src2, dst2, hw2, dinv2, g2, N_EDGES);
    relu_concat2<<<ez(N_NODES * 32), 256, 0, stream>>>(g1, g2, x12b, nullptr, N_NODES);

    fill_f32<<<ez(NG * 64), 256, 0, stream>>>(pooled, 0.0f, NG * 64);
    fill_f32<<<ez(NG), 256, 0, stream>>>(cnt, 0.0f, NG);
    pool_scatter<<<ez(N_NODES * 64), 256, 0, stream>>>(x12b, batch, pooled, cnt, N_NODES);
    pool_div<<<ez(NG * 64), 256, 0, stream>>>(pooled, cnt);
    cvt_f16<<<ez(NG * 64), 256, 0, stream>>>(pooled, pooled16, NG * 64);
    launch_gemm(pooled16, fcxdT, fc1_xd_b, xd, xd16, NG, 128, 64, 64, 128, 0, 0, 1, 1, stream);

    // ---------------- protein branch ----------------
    protein_bins<<<NG, 64, 0, stream>>>(wtr, target, convxt_b, Wsum16);
    // xt[b] = Wsum[b] (64x32) @ embT (32->512), bias folded into slot 31
    launch_gemm(Wsum16, embpT, nullptr, nullptr, xt51216, 64, EMBD, 32, 32, EMBD,
                (long)64 * 32, (long)64 * EMBD, 0, NG, stream);
    launch_gemm(xt51216, encfcT, enc_fc_b, hA, nullptr, 64, 64, 512, 512, 64,
                (long)64 * EMBD, (long)64 * 64, 0, NG, stream);

    conv_glu_layer<<<NG, 256, 0, stream>>>(hA, cwT16, enc_conv_b, hB, 0);
    conv_glu_layer<<<NG, 256, 0, stream>>>(hB, cwT16, enc_conv_b, hA, 1);
    conv_glu_layer<<<NG, 256, 0, stream>>>(hA, cwT16, enc_conv_b, hB, 2);
    conv_glu_layer<<<NG, 256, 0, stream>>>(hB, cwT16, enc_conv_b, hA, 3);

    layernorm64<<<ez(NG * 64), 256, 0, stream>>>(hA, ln_g, ln_b, ln16);
    launch_gemm(ln16, fcxtT, fc1_xt_b, xt128, xt12816, NG, 128, 4096, 4096, 128, 0, 0, 0, 1, stream);

    // ---------------- attention ----------------
    launch_gemm(xd16,    fqT, fq_b, Qb, nullptr, NG, 128, 128, 128, 128, 0, 0, 0, 1, stream);
    launch_gemm(xt12816, fkT, fk_b, Kb, nullptr, NG, 128, 128, 128, 128, 0, 0, 0, 1, stream);
    launch_gemm(xt12816, fvT, fv_b, Vb, nullptr, NG, 128, 128, 128, 128, 0, 0, 0, 1, stream);
    attention_rank1<<<(NG * 128 * 32) / 256, 256, 0, stream>>>(Qb, Kb, Vb, att16);
    launch_gemm(att16, foT, fo_b, atto, nullptr, NG, 128, 128, 128, 128, 0, 0, 0, 1, stream);

    // ---------------- MLP tail ----------------
    concat3<<<ez(NG * 384), 256, 0, stream>>>(xd, atto, xt128, f38416);
    launch_gemm(f38416, m1T, m1_b, nullptr, h116, NG, 1024, 384, 384, 1024, 0, 0, 1, 1, stream);
    launch_gemm(h116,   m2T, m2_b, nullptr, h216, NG, 1024, 1024, 1024, 1024, 0, 0, 1, 1, stream);
    launch_gemm(h216,   m3T, m3_b, h3, nullptr, NG, 256, 1024, 1024, 256, 0, 0, 1, 1, stream);
    final_m4<<<ez(NG), 256, 0, stream>>>(h3, m4_w, m4_b, out);
}
